// VQLiteCodec_71597104825035
// MI455X (gfx1250) — matrix operand order repaired and checked by reading, not compiled, not run on hardware
//
#include <hip/hip_runtime.h>

// ---------------------------------------------------------------------------
// VQ codec for MI455X (gfx1250, wave32, WMMA + async global->LDS).
// dist(n,k) = ||x||^2 + ||c_k||^2 - 2 x.c_k ; argmin over k is independent of
// ||x||^2 so we only evaluate c2[k] - 2*(x.c_k). The matmul runs as bf16x3
// fp32-emulation on v_wmma_f32_16x16x32_bf16 (hi*hi + hi*lo + lo*hi), fp32
// accumulate. Codebook tiles are double-buffered in LDS and prefetched with
// GLOBAL_LOAD_ASYNC_TO_LDS_B128 (ASYNCcnt) so the L2->LDS stream overlaps the
// WMMA work. Compute-bound problem (1.4e11 FLOP vs ~72MB traffic).
// ---------------------------------------------------------------------------

#define DI   256   // embedding dim
#define LDP  264   // padded LDS row stride (ushorts): 528B row -> conflict-free b128
#define BM   64    // query rows per block
#define BN   64    // codebook rows per K-tile

typedef unsigned short ushort_t;
typedef unsigned int   uint_t;
typedef __attribute__((ext_vector_type(16))) __bf16 v16bf;
typedef __attribute__((ext_vector_type(8)))  float  v8f;

__device__ __forceinline__ ushort_t f2bf(float f) {
  uint_t u = __float_as_uint(f);
  u += 0x7FFFu + ((u >> 16) & 1u);      // round-to-nearest-even to bf16
  return (ushort_t)(u >> 16);
}
__device__ __forceinline__ float bf2f(ushort_t h) {
  return __uint_as_float(((uint_t)h) << 16);
}

// Async DMA of 16B global -> LDS (no VGPR data path, tracked by ASYNCcnt).
__device__ __forceinline__ void async_g2l_b128(void* lds_ptr, const void* gptr) {
  const uint_t lds_off = (uint_t)(uintptr_t)lds_ptr;   // low 32b of generic = LDS offset
  asm volatile("global_load_async_to_lds_b128 %0, %1, off"
               :: "v"(lds_off), "v"(gptr)
               : "memory");
}
__device__ __forceinline__ void wait_async_zero() {
  asm volatile("s_wait_asynccnt 0x0" ::: "memory");
}

// ---------------------------------------------------------------------------
// Prep: codebook fp32 -> (hi, lo) bf16 planes + per-code sum of squares.
// One block (256 thr) per codebook row.
// ---------------------------------------------------------------------------
__global__ void vq_prep_codebook(const float* __restrict__ cb,
                                 ushort_t* __restrict__ cb_hi,
                                 ushort_t* __restrict__ cb_lo,
                                 float* __restrict__ c2) {
  const int k = blockIdx.x;
  const int d = threadIdx.x;                 // 256 threads == DI
  const size_t off = (size_t)k * DI + d;
  float v = cb[off];
  ushort_t hi = f2bf(v);
  cb_hi[off] = hi;
  cb_lo[off] = f2bf(v - bf2f(hi));

  float s = v * v;
  for (int o = 16; o >= 1; o >>= 1) s += __shfl_xor(s, o, 32);
  __shared__ float ws[8];
  if ((threadIdx.x & 31) == 0) ws[threadIdx.x >> 5] = s;
  __syncthreads();
  if (threadIdx.x == 0) {
    float t = 0.f;
    #pragma unroll
    for (int i = 0; i < 8; ++i) t += ws[i];
    c2[k] = t;
  }
}

// ---------------------------------------------------------------------------
// Main: per block of BM=64 query rows, sweep all K codes in BN=64 tiles.
// 8 waves: 4 M-waves x 2 N-waves; each wave owns two 16x16 f32 accumulators.
// ---------------------------------------------------------------------------
__global__ __launch_bounds__(256)
void vq_argmin_gather(const float* __restrict__ h,
                      const float* __restrict__ cb,
                      const ushort_t* __restrict__ cb_hi,
                      const ushort_t* __restrict__ cb_lo,
                      const float* __restrict__ c2,
                      float* __restrict__ q_out,
                      float* __restrict__ idx_out,
                      int Kcodes) {
  __shared__ ushort_t Ah[BM * LDP];
  __shared__ ushort_t Al[BM * LDP];
  __shared__ ushort_t Bh[2][BN * LDP];      // double-buffered codebook tile
  __shared__ ushort_t Bl[2][BN * LDP];
  __shared__ float redV[2 * BM];
  __shared__ int   redI[2 * BM];
  __shared__ int   bestI[BM];

  const int tid  = threadIdx.x;
  const int lane = tid & 31;
  const int wav  = tid >> 5;
  const int mw   = wav & 3;          // M-wave: rows mw*16 .. +15
  const int nw   = wav >> 2;         // N-wave: tile cols nw*32 .. +31
  const int half = lane >> 4;        // lane half (ISA fragment layout)
  const int l15  = lane & 15;
  const int r0   = blockIdx.x * BM;

  // Kick off the async DMA for codebook tile 0 before touching h.
  {
    for (int i = tid; i < BN * (DI / 8); i += 256) {
      const int row = i >> 5;                 // DI/8 == 32 chunks per row
      const int ch  = i & 31;
      const size_t goff = (size_t)row * DI + (size_t)ch * 8;
      async_g2l_b128(&Bh[0][row * LDP + ch * 8], cb_hi + goff);
      async_g2l_b128(&Bl[0][row * LDP + ch * 8], cb_lo + goff);
    }
  }

  // Stage h tile -> bf16 hi/lo in LDS (coalesced: col = tid).
  for (int j = 0; j < BM; ++j) {
    float v = h[(size_t)(r0 + j) * DI + tid];
    ushort_t hi = f2bf(v);
    Ah[j * LDP + tid] = hi;
    Al[j * LDP + tid] = f2bf(v - bf2f(hi));
  }

  float minv[8];
  int   mini[8];
  #pragma unroll
  for (int g = 0; g < 8; ++g) { minv[g] = 3.4e38f; mini[g] = 0; }

  const int arow  = (mw * 16 + l15) * LDP;
  const int brow0 = (nw * 32 + l15) * LDP;
  const int brow1 = (nw * 32 + 16 + l15) * LDP;

  int buf = 0;
  for (int kt = 0; kt < Kcodes; kt += BN) {
    wait_async_zero();   // this wave's DMA for tile kt has landed
    __syncthreads();     // => every wave's DMA landed; tile kt-1 compute done

    // Prefetch next tile into the other buffer while we compute this one.
    // Safe: the buffer being refilled was last read during tile kt-1, and the
    // barrier above proves all waves finished that compute.
    if (kt + BN < Kcodes) {
      const int nb = buf ^ 1;
      for (int i = tid; i < BN * (DI / 8); i += 256) {
        const int row = i >> 5;
        const int ch  = i & 31;
        const size_t goff = (size_t)(kt + BN + row) * DI + (size_t)ch * 8;
        async_g2l_b128(&Bh[nb][row * LDP + ch * 8], cb_hi + goff);
        async_g2l_b128(&Bl[nb][row * LDP + ch * 8], cb_lo + goff);
      }
    }

    const ushort_t* bhp = Bh[buf];
    const ushort_t* blp = Bl[buf];

    v8f acc0 = {};
    v8f acc1 = {};

    for (int kk = 0; kk < DI; kk += 32) {
      v16bf a_hi, a_lo, b0h, b0l, b1h, b1l;
      // A fragment: lanes 0-15 hold K{0..7,16..23}, lanes 16-31 K{8..15,24..31}.
      {
        const ushort_t* p = Ah + arow + kk + half * 8;
        ((uint4*)&a_hi)[0] = *(const uint4*)(p);
        ((uint4*)&a_hi)[1] = *(const uint4*)(p + 16);
        const ushort_t* q = Al + arow + kk + half * 8;
        ((uint4*)&a_lo)[0] = *(const uint4*)(q);
        ((uint4*)&a_lo)[1] = *(const uint4*)(q + 16);
      }
      // B fragment: B = cb^T, lane n reads code row n at K{0..7,16..23} / K{8..15,24..31}.
      {
        const ushort_t* p = bhp + brow0 + kk + half * 8;
        ((uint4*)&b0h)[0] = *(const uint4*)(p);
        ((uint4*)&b0h)[1] = *(const uint4*)(p + 16);
        const ushort_t* q = blp + brow0 + kk + half * 8;
        ((uint4*)&b0l)[0] = *(const uint4*)(q);
        ((uint4*)&b0l)[1] = *(const uint4*)(q + 16);
      }
      {
        const ushort_t* p = bhp + brow1 + kk + half * 8;
        ((uint4*)&b1h)[0] = *(const uint4*)(p);
        ((uint4*)&b1h)[1] = *(const uint4*)(p + 16);
        const ushort_t* q = blp + brow1 + kk + half * 8;
        ((uint4*)&b1l)[0] = *(const uint4*)(q);
        ((uint4*)&b1l)[1] = *(const uint4*)(q + 16);
      }
      // bf16x3 fp32 emulation: hi*hi + hi*lo + lo*hi, fp32 accumulate.
      acc0 = __builtin_amdgcn_wmma_f32_16x16x32_bf16(false, a_hi, false, b0h, (short)0, acc0, false, false);
      acc0 = __builtin_amdgcn_wmma_f32_16x16x32_bf16(false, a_hi, false, b0l, (short)0, acc0, false, false);
      acc0 = __builtin_amdgcn_wmma_f32_16x16x32_bf16(false, a_lo, false, b0h, (short)0, acc0, false, false);
      acc1 = __builtin_amdgcn_wmma_f32_16x16x32_bf16(false, a_hi, false, b1h, (short)0, acc1, false, false);
      acc1 = __builtin_amdgcn_wmma_f32_16x16x32_bf16(false, a_hi, false, b1l, (short)0, acc1, false, false);
      acc1 = __builtin_amdgcn_wmma_f32_16x16x32_bf16(false, a_lo, false, b1h, (short)0, acc1, false, false);
    }

    // Fold tile into running argmin: compare c2[n] - 2*xc (||x||^2 irrelevant).
    const int n0 = kt + nw * 32 + l15;
    const float c2a = c2[n0];
    const float c2b = c2[n0 + 16];
    #pragma unroll
    for (int g = 0; g < 8; ++g) {
      float d0 = c2a - 2.0f * acc0[g];
      float d1 = c2b - 2.0f * acc1[g];
      if (d0 < minv[g]) { minv[g] = d0; mini[g] = n0; }       // strict < keeps
      if (d1 < minv[g]) { minv[g] = d1; mini[g] = n0 + 16; }  // first-min idx
    }
    buf ^= 1;
  }

  // Reduce across the 16 lanes that share a row (C layout: N = lane%16).
  #pragma unroll
  for (int g = 0; g < 8; ++g) {
    float v = minv[g];
    int   id = mini[g];
    for (int o = 8; o >= 1; o >>= 1) {
      float ov = __shfl_xor(v, o, 32);
      int   oi = __shfl_xor(id, o, 32);
      if (ov < v || (ov == v && oi < id)) { v = ov; id = oi; }
    }
    if (l15 == 0) {
      const int rowL = mw * 16 + half * 8 + g;   // C layout: VGPR g -> M=g / g+8
      redV[nw * BM + rowL] = v;
      redI[nw * BM + rowL] = id;
    }
  }
  __syncthreads();

  // Merge the two N-wave halves; emit index (as float value).
  if (tid < BM) {
    float v0 = redV[tid];       int i0 = redI[tid];
    float v1 = redV[BM + tid];  int i1 = redI[BM + tid];
    int sel = (v1 < v0 || (v1 == v0 && i1 < i0)) ? i1 : i0;
    bestI[tid] = sel;
    idx_out[r0 + tid] = (float)sel;
  }
  __syncthreads();

  // Gather q = codebook[best] in fp32 (STE forward value == q). col = tid.
  for (int j = 0; j < BM; ++j) {
    const int code = bestI[j];
    q_out[(size_t)(r0 + j) * DI + tid] = cb[(size_t)code * DI + tid];
  }
}

// ---------------------------------------------------------------------------
extern "C" void kernel_launch(void* const* d_in, const int* in_sizes, int n_in,
                              void* d_out, int out_size, void* d_ws, size_t ws_size,
                              hipStream_t stream) {
  const float* h  = (const float*)d_in[0];      // [B*T, D] fp32
  const float* cb = (const float*)d_in[1];      // [K, D]   fp32
  const int N  = in_sizes[0] / DI;              // 32768
  const int Kc = in_sizes[1] / DI;              // 8192

  // Workspace: cb_hi (K*D u16) | cb_lo (K*D u16) | c2 (K f32)  ~= 8 MB
  ushort_t* cb_hi = (ushort_t*)d_ws;
  ushort_t* cb_lo = cb_hi + (size_t)Kc * DI;
  float*    c2    = (float*)(cb_lo + (size_t)Kc * DI);

  float* q_out   = (float*)d_out;               // N*D floats
  float* idx_out = q_out + (size_t)N * DI;      // N entries (written as float)

  vq_prep_codebook<<<Kc, 256, 0, stream>>>(cb, cb_hi, cb_lo, c2);
  vq_argmin_gather<<<N / BM, 256, 0, stream>>>(h, cb, cb_hi, cb_lo, c2,
                                               q_out, idx_out, Kc);
}
